// MultiHeadAttention_18923625906347
// MI455X (gfx1250) — matrix operand order repaired and checked by reading, not compiled, not run on hardware
//
#include <hip/hip_runtime.h>
#include <hip/hip_bf16.h>

// ---------------------------------------------------------------------------
// MHA forward for MI455X (gfx1250, wave32, WMMA).
// Pass 1..3: Q/K/V projections (f32 -> bf16, head-split [B,H,S,HD]),
//            LDS-staged double-buffered NT GEMM with async-to-LDS copies
//            (ASYNCcnt / global_load_async_to_lds_b128).
//            SCALE (1/sqrt(HD)) folded into the Q projection epilogue.
// Pass 4   : flash attention, Bc=64 key tile, online softmax, V^T in LDS.
// Pass 5   : output projection (bf16 -> f32 into d_out).
// All matmuls: v_wmma_f32_16x16x32_bf16 (f32 accumulate).
// ---------------------------------------------------------------------------

typedef __bf16 bf16_t;
typedef __attribute__((ext_vector_type(16))) __bf16 v16bf;
typedef __attribute__((ext_vector_type(8)))  __bf16 v8bf;
typedef __attribute__((ext_vector_type(8)))  float  v8f;
typedef __attribute__((ext_vector_type(4)))  float  v4f;
typedef int v4i_gcc __attribute__((__vector_size__(16)));

#define B_  4
#define S_  2048
#define E_  2048
#define H_  16
#define HD_ 128
#define M_  (B_ * S_)                       // 8192 rows
#define SCALE_ 0.08838834764831845f        // 1/sqrt(128)

#ifndef __has_builtin
#define __has_builtin(x) 0
#endif

// Async global->LDS copy path (ASYNCcnt). Confirmed on this toolchain:
// (AS1 v4i* src, AS3 v4i* dst, imm offset, imm cpol).
#if __has_builtin(__builtin_amdgcn_global_load_async_to_lds_b128)
#define ASYNC_COPY 1
#define COPY16(g, l)                                                      \
  __builtin_amdgcn_global_load_async_to_lds_b128(                         \
      (__attribute__((address_space(1))) v4i_gcc*)(g),                    \
      (__attribute__((address_space(3))) v4i_gcc*)(l), 0, 0)
#if __has_builtin(__builtin_amdgcn_s_wait_asynccnt)
#define WAIT_ASYNC(n) __builtin_amdgcn_s_wait_asynccnt(n)
#else
#define WAIT_ASYNC(n) asm volatile("s_wait_asynccnt %0" ::"i"(n) : "memory")
#endif
#else
#define ASYNC_COPY 0
#define COPY16(g, l) (*(v4f*)(l) = *(const v4f*)(g))
#define WAIT_ASYNC(n)
#endif

__device__ __forceinline__ v16bf cat16(v8bf lo, v8bf hi) {
  return __builtin_shufflevector(lo, hi, 0,1,2,3,4,5,6,7,8,9,10,11,12,13,14,15);
}

// load 8 contiguous f32 (16B-aligned, global or LDS) and convert to bf16
__device__ __forceinline__ v8bf cvt8(const float* p) {
  const v4f* q = (const v4f*)p;
  v4f a = q[0], b = q[1];
  v8bf r;
  r[0]=(bf16_t)a[0]; r[1]=(bf16_t)a[1]; r[2]=(bf16_t)a[2]; r[3]=(bf16_t)a[3];
  r[4]=(bf16_t)b[0]; r[5]=(bf16_t)b[1]; r[6]=(bf16_t)b[2]; r[7]=(bf16_t)b[3];
  return r;
}

__device__ __forceinline__ v8f wmma_bf16(v16bf a, v16bf b, v8f c) {
  // (neg_a, A, neg_b, B, c_mod, C, reuse_a, reuse_b)
  return __builtin_amdgcn_wmma_f32_16x16x32_bf16(false, a, false, b, (short)0, c,
                                                 false, false);
}

__device__ __forceinline__ v8f vzero() {
  v8f z = {0.f,0.f,0.f,0.f,0.f,0.f,0.f,0.f};
  return z;
}

__device__ __forceinline__ void wave_lds_sync() {
  __builtin_amdgcn_wave_barrier();
  asm volatile("s_wait_dscnt 0" ::: "memory");
  __builtin_amdgcn_wave_barrier();
}

// ---------------------------------------------------------------------------
// NT GEMM: Y[m,n] = (sum_k X[m,k]*W[n,k] + bias[n]) * cscale
// 256 threads = 8 waves (4 m x 2 n), block tile 128x128, wave tile 32x64,
// K-step 32, double-buffered LDS tiles (X: f32 or bf16; W: f32).
// ---------------------------------------------------------------------------
template <bool IN_BF16, bool OUT_F32>
__global__ __launch_bounds__(256)
void gemm_nt_kernel(const void* __restrict__ Xv, const float* __restrict__ W,
                    const float* __restrict__ bias, void* __restrict__ Y,
                    float cscale)
{
  constexpr int ESZ    = IN_BF16 ? 2 : 4;
  constexpr int XBYTES = 128 * 32 * ESZ;
  constexpr int WBYTES = 128 * 32 * 4;
  constexpr int XCH    = XBYTES / 16;            // 16B chunks per X tile
  constexpr int WCH    = WBYTES / 16;            // 1024
  constexpr int XCPR   = (32 * ESZ) / 16;        // chunks per X row
  constexpr int PER_STEP = XCH / 256 + WCH / 256;  // async ops per thread/step

  __shared__ __align__(16) char sX[2][XBYTES];
  __shared__ __align__(16) char sW[2][WBYTES];

  const int tid  = threadIdx.x;
  const int lane = tid & 31, wid = tid >> 5;
  const int ln = lane & 15, hl = lane >> 4;
  const int wm = wid & 3,  wn = wid >> 2;
  const int m0b = blockIdx.x * 128;
  const int n0b = blockIdx.y * 128;

  const char* Xc = (const char*)Xv;
  const char* Wc = (const char*)W;

  auto stage = [&](int buf, int k0) {
#pragma unroll
    for (int i = 0; i < XCH / 256; ++i) {
      const int c = tid + i * 256;
      const int row = c / XCPR, kc = c % XCPR;
      const char* g = Xc + ((size_t)(m0b + row) * E_ + k0) * ESZ + kc * 16;
      COPY16(g, &sX[buf][c * 16]);
    }
#pragma unroll
    for (int i = 0; i < WCH / 256; ++i) {
      const int c = tid + i * 256;
      const int row = c >> 3, kc = c & 7;
      const char* g = Wc + ((size_t)(n0b + row) * E_ + k0) * 4 + kc * 16;
      COPY16(g, &sW[buf][c * 16]);
    }
  };

  v8f acc[2][4];
#pragma unroll
  for (int mi = 0; mi < 2; ++mi)
#pragma unroll
    for (int ni = 0; ni < 4; ++ni) acc[mi][ni] = vzero();

  stage(0, 0);
  constexpr int NIT = E_ / 32;
  for (int it = 0; it < NIT; ++it) {
    const int cur = it & 1;
    if (it + 1 < NIT) stage(cur ^ 1, (it + 1) * 32);
#if ASYNC_COPY
    if (it + 1 < NIT) { WAIT_ASYNC(PER_STEP); } else { WAIT_ASYNC(0); }
#endif
    __syncthreads();

    v16bf a[2], bfr[4];
#pragma unroll
    for (int mi = 0; mi < 2; ++mi) {
      const int row = wm * 32 + mi * 16 + ln;
      if (IN_BF16) {
        const bf16_t* xb = (const bf16_t*)&sX[cur][0] + row * 32;
        a[mi] = cat16(*(const v8bf*)(xb + hl * 8),
                      *(const v8bf*)(xb + 16 + hl * 8));
      } else {
        const float* xb = (const float*)&sX[cur][0] + row * 32;
        a[mi] = cat16(cvt8(xb + hl * 8), cvt8(xb + 16 + hl * 8));
      }
    }
#pragma unroll
    for (int ni = 0; ni < 4; ++ni) {
      const float* wb =
          (const float*)&sW[cur][0] + (wn * 64 + ni * 16 + ln) * 32;
      bfr[ni] = cat16(cvt8(wb + hl * 8), cvt8(wb + 16 + hl * 8));
    }
#pragma unroll
    for (int mi = 0; mi < 2; ++mi)
#pragma unroll
      for (int ni = 0; ni < 4; ++ni)
        acc[mi][ni] = wmma_bf16(a[mi], bfr[ni], acc[mi][ni]);
    __syncthreads();
  }

  // epilogue: bias + scale + store. C layout: VGPR r -> row r+8*hl, col = ln.
#pragma unroll
  for (int ni = 0; ni < 4; ++ni) {
    const int col = n0b + wn * 64 + ni * 16 + ln;
    const float bv = bias[col];
#pragma unroll
    for (int mi = 0; mi < 2; ++mi) {
#pragma unroll
      for (int r = 0; r < 8; ++r) {
        const int row = m0b + wm * 32 + mi * 16 + r + 8 * hl;
        const float val = (acc[mi][ni][r] + bv) * cscale;
        if (OUT_F32) {
          ((float*)Y)[(size_t)row * E_ + col] = val;
        } else {
          const int bb = row >> 11, ss = row & (S_ - 1);
          const int hh = col >> 7,  dd = col & (HD_ - 1);
          ((bf16_t*)Y)[(((size_t)(bb * H_ + hh)) * S_ + ss) * HD_ + dd] =
              (bf16_t)val;
        }
      }
    }
  }
}

// ---------------------------------------------------------------------------
// Flash attention: grid (S/64, H, B), 128 threads = 4 waves, each wave owns
// 16 query rows. Keys streamed 64 at a time (Bc=64); online softmax with
// cross-lane reductions in 16-lane groups; V staged transposed in LDS so
// P@V B-fragments are contiguous ds_load_b128. SCALE pre-folded into Q.
// ---------------------------------------------------------------------------
__global__ __launch_bounds__(128)
void flash_attn_kernel(const bf16_t* __restrict__ Q, const bf16_t* __restrict__ K,
                       const bf16_t* __restrict__ V, bf16_t* __restrict__ Ctx)
{
  __shared__ __align__(16) bf16_t sVt[HD_ * 64];   // V^T tile [d][key]  16KB
  __shared__ __align__(16) bf16_t sP[4][16 * 64];  // per-wave P tile     8KB

  const int tid  = threadIdx.x;
  const int lane = tid & 31, wid = tid >> 5;
  const int ln = lane & 15, hl = lane >> 4;
  const int qb = blockIdx.x, h = blockIdx.y, b = blockIdx.z;
  const size_t off = (((size_t)b * H_ + h) * S_) * HD_;
  const int q0 = qb * 64 + wid * 16;

  // Q A-fragments over 4 k-steps of 32 (HD=128), kept in registers
  v16bf qf[4];
#pragma unroll
  for (int kd = 0; kd < 4; ++kd) {
    const bf16_t* p = Q + off + (size_t)(q0 + ln) * HD_ + kd * 32;
    qf[kd] = cat16(*(const v8bf*)(p + hl * 8), *(const v8bf*)(p + 16 + hl * 8));
  }

  v8f o[8];
#pragma unroll
  for (int df = 0; df < 8; ++df) o[df] = vzero();
  float mr[8], lr[8];
#pragma unroll
  for (int r = 0; r < 8; ++r) { mr[r] = -1e30f; lr[r] = 0.f; }

  const bf16_t* Kp = K + off;
  const bf16_t* Vp = V + off;
  bf16_t* myP = &sP[wid][0];

  for (int kb = 0; kb < S_; kb += 64) {
    __syncthreads();
    // stage V^T: thread tid owns d=tid, walks 64 keys (coalesced reads)
    for (int i = 0; i < 64; ++i)
      sVt[tid * 64 + i] = Vp[(size_t)(kb + i) * HD_ + tid];
    __syncthreads();

    // scores: 16 queries x 64 keys, f32 accumulate (Q already carries SCALE)
    v8f sc[4];
#pragma unroll
    for (int nf = 0; nf < 4; ++nf) sc[nf] = vzero();
#pragma unroll
    for (int kd = 0; kd < 4; ++kd)
#pragma unroll
      for (int nf = 0; nf < 4; ++nf) {
        const bf16_t* p =
            Kp + (size_t)(kb + nf * 16 + ln) * HD_ + kd * 32;
        sc[nf] = wmma_bf16(qf[kd],
                           cat16(*(const v8bf*)(p + hl * 8), *(const v8bf*)(p + 16 + hl * 8)),
                           sc[nf]);
      }

    // online softmax; row r+8*hl lives in VGPR r, cols across 16 lanes
#pragma unroll
    for (int r = 0; r < 8; ++r) {
      const float s0 = sc[0][r], s1 = sc[1][r], s2 = sc[2][r], s3 = sc[3][r];
      float mx = fmaxf(fmaxf(s0, s1), fmaxf(s2, s3));
      mx = fmaxf(mx, __shfl_xor(mx, 1, 16));
      mx = fmaxf(mx, __shfl_xor(mx, 2, 16));
      mx = fmaxf(mx, __shfl_xor(mx, 4, 16));
      mx = fmaxf(mx, __shfl_xor(mx, 8, 16));
      const float mnew = fmaxf(mr[r], mx);
      const float p0 = __expf(s0 - mnew), p1 = __expf(s1 - mnew);
      const float p2 = __expf(s2 - mnew), p3 = __expf(s3 - mnew);
      float sum = (p0 + p1) + (p2 + p3);
      sum += __shfl_xor(sum, 1, 16);
      sum += __shfl_xor(sum, 2, 16);
      sum += __shfl_xor(sum, 4, 16);
      sum += __shfl_xor(sum, 8, 16);
      const float alpha = __expf(mr[r] - mnew);
      lr[r] = lr[r] * alpha + sum;
      mr[r] = mnew;
#pragma unroll
      for (int df = 0; df < 8; ++df) o[df][r] *= alpha;
      const int prow = (r + 8 * hl) * 64;
      myP[prow + ln]      = (bf16_t)p0;
      myP[prow + 16 + ln] = (bf16_t)p1;
      myP[prow + 32 + ln] = (bf16_t)p2;
      myP[prow + 48 + ln] = (bf16_t)p3;
    }
    wave_lds_sync();  // C-layout -> A-layout relayout is cross-lane via LDS

    // P as two 16x32 A-fragments
    v16bf pa[2];
#pragma unroll
    for (int kk = 0; kk < 2; ++kk) {
      const bf16_t* pp = myP + ln * 64 + kk * 32 + hl * 8;
      pa[kk] = cat16(*(const v8bf*)pp, *(const v8bf*)(pp + 16));
    }

    // o += P @ V : B-frag lane reads V^T[d=n][keys 8h..8h+7, 16+8h..16+8h+7]
#pragma unroll
    for (int df = 0; df < 8; ++df)
#pragma unroll
      for (int kk = 0; kk < 2; ++kk) {
        const bf16_t* vp = sVt + (df * 16 + ln) * 64 + kk * 32;
        o[df] = wmma_bf16(pa[kk],
                          cat16(*(const v8bf*)(vp + hl * 8), *(const v8bf*)(vp + 16 + hl * 8)),
                          o[df]);
      }
  }

  // epilogue: normalize and store ctx in [B,S,E] (head-interleaved)
#pragma unroll
  for (int r = 0; r < 8; ++r) {
    const float inv = 1.0f / lr[r];
    const int srow = q0 + r + 8 * hl;
    const size_t base = ((size_t)b * S_ + srow) * E_ + (size_t)h * HD_;
#pragma unroll
    for (int df = 0; df < 8; ++df)
      Ctx[base + df * 16 + ln] = (bf16_t)(o[df][r] * inv);
  }
}

// ---------------------------------------------------------------------------
extern "C" void kernel_launch(void* const* d_in, const int* in_sizes, int n_in,
                              void* d_out, int out_size, void* d_ws, size_t ws_size,
                              hipStream_t stream) {
  (void)in_sizes; (void)n_in; (void)out_size; (void)ws_size;
  const float* QKV = (const float*)d_in[0];
  const float* Wq  = (const float*)d_in[1];
  const float* bq  = (const float*)d_in[2];
  const float* Wk  = (const float*)d_in[3];
  const float* bk  = (const float*)d_in[4];
  const float* Wv  = (const float*)d_in[5];
  const float* bv  = (const float*)d_in[6];
  const float* Wo  = (const float*)d_in[7];
  const float* bo  = (const float*)d_in[8];
  // d_in[9] = is_causal: mask is a no-op in the reference, ignored.

  const size_t N = (size_t)B_ * S_ * E_;       // 16 Mi elements
  bf16_t* Qb = (bf16_t*)d_ws;                  // [B,H,S,HD] bf16 (pre-scaled)
  bf16_t* Kb = Qb + N;
  bf16_t* Vb = Kb + N;
  bf16_t* Cb = Vb + N;                         // ctx [B,S,E] bf16

  dim3 gg(M_ / 128, E_ / 128);
  gemm_nt_kernel<false, false><<<gg, 256, 0, stream>>>(QKV, Wq, bq, Qb, SCALE_);
  gemm_nt_kernel<false, false><<<gg, 256, 0, stream>>>(QKV, Wk, bk, Kb, 1.0f);
  gemm_nt_kernel<false, false><<<gg, 256, 0, stream>>>(QKV, Wv, bv, Vb, 1.0f);

  dim3 fg(S_ / 64, H_, B_);
  flash_attn_kernel<<<fg, 128, 0, stream>>>(Qb, Kb, Vb, Cb);

  gemm_nt_kernel<true, true><<<gg, 256, 0, stream>>>(Cb, Wo, bo, d_out, 1.0f);
}
